// ReXMoEInferenceMLP_5205500362822
// MI455X (gfx1250) — matrix operand order repaired and checked by reading, not compiled, not run on hardware
//
#include <hip/hip_runtime.h>

// ---------------------------------------------------------------------------
// ReXMoE inference MLP for MI455X (gfx1250, wave32, WMMA bf16).
// Pipeline: fp32->bf16 convert, router(top-2)+compaction, base SwiGLU GEMMs,
// sparse expert SwiGLU GEMMs (gathered rows), final mix. All matmuls use
// v_wmma_f32_16x16x32_bf16 with fp32 accumulation. Each GEMM wave owns a
// 32(M) x 64(N) tile: two A fragments, every B fragment feeds two WMMAs.
// ---------------------------------------------------------------------------

typedef __attribute__((ext_vector_type(16))) __bf16 v16bf;
typedef __attribute__((ext_vector_type(8)))  __bf16 v8bf;
typedef __attribute__((ext_vector_type(4)))  __bf16 v4bf;
typedef __attribute__((ext_vector_type(8)))  float  v8f;
typedef __attribute__((ext_vector_type(4)))  float  v4f;

#define T_TOK 2048   // B*S tokens
#define HDIM  2048   // hidden dim
#define FB    8192   // base FFN dim
#define NEXP  8      // experts
#define FE    1024   // expert FFN dim
#define ENT   (2 * T_TOK)  // total routed entries (top_k = 2)

// ---------------------------------------------------------------------------
// WMMA helpers (CDNA5 bf16 16x16x32, fp32 accumulate)
// ---------------------------------------------------------------------------
__device__ __forceinline__ v8f wmma_bf16(v16bf a, v16bf b, v8f c) {
  // (neg_a, A, neg_b, B, c_mod, C, reuse_a, reuse_b)
  return __builtin_amdgcn_wmma_f32_16x16x32_bf16(false, a, false, b,
                                                 (short)0, c, false, false);
}

// A fragment (16x32 bf16): lane row is chosen by the caller (row pointer).
// lanes 0-15 hold K = k0..k0+15 (contiguous), lanes 16-31 shifted by +16.
__device__ __forceinline__ v16bf load_a_frag(const __bf16* row, int k0, int lane) {
  const int kb = k0 + ((lane >> 4) << 4);
  v8bf lo = *(const v8bf*)(row + kb);
  v8bf hi = *(const v8bf*)(row + kb + 8);
  v16bf r;
#pragma unroll
  for (int i = 0; i < 8; ++i) { r[i] = lo[i]; r[i + 8] = hi[i]; }
  return r;
}

// B fragment (32x16 bf16) for C = A * W^T: column n of B == row n of W.
// lane l holds W[n0+(l&15)][k0 + (l>=16 ? 16 : 0) .. +15] (contiguous).
__device__ __forceinline__ v16bf load_b_frag(const __bf16* w, int ld, int n0,
                                             int k0, int lane) {
  const int n  = n0 + (lane & 15);
  const int kb = k0 + ((lane >> 4) << 4);
  return *(const v16bf*)(w + (long)n * ld + kb);
}

__device__ __forceinline__ float silu_f(float g) {
  return g / (1.0f + __expf(-g));
}

// ---------------------------------------------------------------------------
// fp32 -> bf16 conversion (vectorized x4, grid-stride)
// ---------------------------------------------------------------------------
__global__ __launch_bounds__(256) void cvt_f32_bf16(const float* __restrict__ s,
                                                    __bf16* __restrict__ d,
                                                    long n4) {
  const long stride = (long)gridDim.x * blockDim.x;
  for (long i = (long)blockIdx.x * blockDim.x + threadIdx.x; i < n4; i += stride) {
    v4f v = *(const v4f*)(s + i * 4);
    v4bf o;
#pragma unroll
    for (int j = 0; j < 4; ++j) o[j] = (__bf16)v[j];
    *(v4bf*)(d + i * 4) = o;
  }
}

// ---------------------------------------------------------------------------
// Router: one wave per token. 8 logits -> top-2 -> softmax -> *alpha.
// ---------------------------------------------------------------------------
__global__ __launch_bounds__(256) void router_kernel(
    const float* __restrict__ x, const float* __restrict__ rw,
    const float* __restrict__ alpha, int* __restrict__ idx0,
    int* __restrict__ idx1, float* __restrict__ w0a, float* __restrict__ w1a,
    float* __restrict__ sarr, int* __restrict__ counts) {
  const int lane = threadIdx.x & 31;
  const int t = blockIdx.x * 8 + (threadIdx.x >> 5);
  const float* xr = x + (long)t * HDIM;
  float logit[NEXP];
#pragma unroll
  for (int e = 0; e < NEXP; ++e) {
    const float* wr = rw + e * HDIM;
    float acc = 0.f;
    for (int i = lane; i < HDIM; i += 32) acc += xr[i] * wr[i];
#pragma unroll
    for (int o = 16; o > 0; o >>= 1) acc += __shfl_xor(acc, o, 32);
    logit[e] = acc;
  }
  if (lane == 0) {
    int i0 = 0; float v0 = logit[0];
#pragma unroll
    for (int e = 1; e < NEXP; ++e)
      if (logit[e] > v0) { v0 = logit[e]; i0 = e; }
    int i1 = (i0 == 0) ? 1 : 0; float v1 = -3.4e38f;
#pragma unroll
    for (int e = 0; e < NEXP; ++e)
      if (e != i0 && logit[e] > v1) { v1 = logit[e]; i1 = e; }
    const float p0 = 1.f / (1.f + __expf(v1 - v0));  // softmax over top-2
    const float p1 = 1.f - p0;
    const float a0 = p0 * alpha[i0];
    const float a1 = p1 * alpha[i1];
    idx0[t] = i0; idx1[t] = i1;
    w0a[t] = a0;  w1a[t] = a1;
    sarr[t] = 1.f - (a0 + a1);  // out = s*base + a0*eo0 + a1*eo1
    atomicAdd(&counts[i0], 1);
    atomicAdd(&counts[i1], 1);
  }
}

__global__ void zero_counts_kernel(int* counts) {
  if (threadIdx.x < NEXP) counts[threadIdx.x] = 0;
}

__global__ void scan_offsets_kernel(const int* __restrict__ counts,
                                    int* __restrict__ offs,
                                    int* __restrict__ cursors) {
  if (threadIdx.x == 0) {
    int run = 0;
    for (int e = 0; e < NEXP; ++e) { offs[e] = run; run += counts[e]; }
  }
  if (threadIdx.x < NEXP) cursors[threadIdx.x] = 0;
}

__global__ __launch_bounds__(256) void assign_slots_kernel(
    const int* __restrict__ idx0, const int* __restrict__ idx1,
    const int* __restrict__ offs, int* __restrict__ cursors,
    int* __restrict__ list, int* __restrict__ slot0, int* __restrict__ slot1) {
  const int t = blockIdx.x * blockDim.x + threadIdx.x;
  if (t >= T_TOK) return;
  const int e0 = idx0[t];
  const int s0 = offs[e0] + atomicAdd(&cursors[e0], 1);
  list[s0] = t; slot0[t] = s0;
  const int e1 = idx1[t];
  const int s1 = offs[e1] + atomicAdd(&cursors[e1], 1);
  list[s1] = t; slot1[t] = s1;
}

// ---------------------------------------------------------------------------
// Base MLP: hidden = silu(x@Wg^T) * (x@Wu^T)  -> bf16 [T, FB]
// Wave tile: 32(M) x 64(N); two A fragments, B reused for both M halves.
// ---------------------------------------------------------------------------
__global__ __launch_bounds__(256) void base_gateup_kernel(
    const __bf16* __restrict__ xb, const __bf16* __restrict__ wg,
    const __bf16* __restrict__ wu, __bf16* __restrict__ hidden) {
  const int lane = threadIdx.x & 31;
  const int wid = blockIdx.x * 8 + (threadIdx.x >> 5);
  const int NG = FB / 64;                       // 128
  const int m0 = (wid / NG) * 32;
  const int n0 = (wid % NG) * 64;
  const __bf16* arow0 = xb + (long)(m0 + (lane & 15)) * HDIM;
  const __bf16* arow1 = xb + (long)(m0 + 16 + (lane & 15)) * HDIM;
  v8f ag[2][4] = {}, au[2][4] = {};
  for (int k0 = 0; k0 < HDIM; k0 += 32) {
    v16bf a0 = load_a_frag(arow0, k0, lane);
    v16bf a1 = load_a_frag(arow1, k0, lane);
#pragma unroll
    for (int j = 0; j < 4; ++j) {
      v16bf bg = load_b_frag(wg, HDIM, n0 + j * 16, k0, lane);
      ag[0][j] = wmma_bf16(a0, bg, ag[0][j]);
      ag[1][j] = wmma_bf16(a1, bg, ag[1][j]);
      v16bf bu = load_b_frag(wu, HDIM, n0 + j * 16, k0, lane);
      au[0][j] = wmma_bf16(a0, bu, au[0][j]);
      au[1][j] = wmma_bf16(a1, bu, au[1][j]);
    }
  }
  const int col = n0 + (lane & 15);
  const int radd = (lane >> 4) * 8;
#pragma unroll
  for (int mi = 0; mi < 2; ++mi)
#pragma unroll
    for (int j = 0; j < 4; ++j)
#pragma unroll
      for (int v = 0; v < 8; ++v) {
        const int row = m0 + mi * 16 + v + radd;
        hidden[(long)row * FB + col + j * 16] =
            (__bf16)(silu_f(ag[mi][j][v]) * au[mi][j][v]);
      }
}

// base = hidden @ Wd^T -> fp32 [T, H]
__global__ __launch_bounds__(256) void base_down_kernel(
    const __bf16* __restrict__ hidden, const __bf16* __restrict__ wd,
    float* __restrict__ baseo) {
  const int lane = threadIdx.x & 31;
  const int wid = blockIdx.x * 8 + (threadIdx.x >> 5);
  const int NG = HDIM / 64;                     // 32
  const int m0 = (wid / NG) * 32;
  const int n0 = (wid % NG) * 64;
  const __bf16* arow0 = hidden + (long)(m0 + (lane & 15)) * FB;
  const __bf16* arow1 = hidden + (long)(m0 + 16 + (lane & 15)) * FB;
  v8f acc[2][4] = {};
  for (int k0 = 0; k0 < FB; k0 += 32) {
    v16bf a0 = load_a_frag(arow0, k0, lane);
    v16bf a1 = load_a_frag(arow1, k0, lane);
#pragma unroll
    for (int j = 0; j < 4; ++j) {
      v16bf b = load_b_frag(wd, FB, n0 + j * 16, k0, lane);
      acc[0][j] = wmma_bf16(a0, b, acc[0][j]);
      acc[1][j] = wmma_bf16(a1, b, acc[1][j]);
    }
  }
  const int col = n0 + (lane & 15);
  const int radd = (lane >> 4) * 8;
#pragma unroll
  for (int mi = 0; mi < 2; ++mi)
#pragma unroll
    for (int j = 0; j < 4; ++j)
#pragma unroll
      for (int v = 0; v < 8; ++v)
        baseo[(long)(m0 + mi * 16 + v + radd) * HDIM + col + j * 16] =
            acc[mi][j][v];
}

// ---------------------------------------------------------------------------
// Sparse expert MLP pass 1: he[slot] = silu(x[tok]@Eg^T) * (x[tok]@Eu^T)
// A rows gathered via per-expert token list; wave-uniform early exit so
// EXEC stays all-ones around WMMA. Row clamp + masked store for tails.
// ---------------------------------------------------------------------------
__global__ __launch_bounds__(256) void expert_gateup_kernel(
    const __bf16* __restrict__ xb, const __bf16* __restrict__ ewg,
    const __bf16* __restrict__ ewu, const int* __restrict__ list,
    const int* __restrict__ cnts, const int* __restrict__ offs,
    __bf16* __restrict__ he) {
  const int lane = threadIdx.x & 31;
  const int wid = blockIdx.x * 8 + (threadIdx.x >> 5);
  const int MT = T_TOK / 32, NG = FE / 64;      // 64, 16
  const int e  = wid / (MT * NG);
  const int mt = (wid / NG) % MT;
  const int n0 = (wid % NG) * 64;
  const int cnt = cnts[e];
  const int m0 = mt * 32;
  if (m0 >= cnt) return;                        // wave-uniform
  const int off = offs[e];
  int r0 = m0 + (lane & 15);
  int r1 = m0 + 16 + (lane & 15);
  if (r0 >= cnt) r0 = cnt - 1;                  // clamp (dup rows, masked store)
  if (r1 >= cnt) r1 = cnt - 1;
  const __bf16* arow0 = xb + (long)list[off + r0] * HDIM;
  const __bf16* arow1 = xb + (long)list[off + r1] * HDIM;
  const __bf16* wg = ewg + (long)e * FE * HDIM;
  const __bf16* wu = ewu + (long)e * FE * HDIM;
  v8f ag[2][4] = {}, au[2][4] = {};
  for (int k0 = 0; k0 < HDIM; k0 += 32) {
    v16bf a0 = load_a_frag(arow0, k0, lane);
    v16bf a1 = load_a_frag(arow1, k0, lane);
#pragma unroll
    for (int j = 0; j < 4; ++j) {
      v16bf bg = load_b_frag(wg, HDIM, n0 + j * 16, k0, lane);
      ag[0][j] = wmma_bf16(a0, bg, ag[0][j]);
      ag[1][j] = wmma_bf16(a1, bg, ag[1][j]);
      v16bf bu = load_b_frag(wu, HDIM, n0 + j * 16, k0, lane);
      au[0][j] = wmma_bf16(a0, bu, au[0][j]);
      au[1][j] = wmma_bf16(a1, bu, au[1][j]);
    }
  }
  const int col = n0 + (lane & 15);
  const int radd = (lane >> 4) * 8;
#pragma unroll
  for (int mi = 0; mi < 2; ++mi)
#pragma unroll
    for (int j = 0; j < 4; ++j)
#pragma unroll
      for (int v = 0; v < 8; ++v) {
        const int m = m0 + mi * 16 + v + radd;
        if (m < cnt)
          he[(long)(off + m) * FE + col + j * 16] =
              (__bf16)(silu_f(ag[mi][j][v]) * au[mi][j][v]);
      }
}

// Expert pass 2: eo[slot] = he[slot] @ Ed^T  -> fp32 [ENT, H]
__global__ __launch_bounds__(256) void expert_down_kernel(
    const __bf16* __restrict__ he, const __bf16* __restrict__ ewd,
    const int* __restrict__ cnts, const int* __restrict__ offs,
    float* __restrict__ eo) {
  const int lane = threadIdx.x & 31;
  const int wid = blockIdx.x * 8 + (threadIdx.x >> 5);
  const int MT = T_TOK / 32, NG = HDIM / 64;    // 64, 32
  const int e  = wid / (MT * NG);
  const int mt = (wid / NG) % MT;
  const int n0 = (wid % NG) * 64;
  const int cnt = cnts[e];
  const int m0 = mt * 32;
  if (m0 >= cnt) return;                        // wave-uniform
  const int off = offs[e];
  // he is padded by 32 rows so reading past cnt inside the tile is safe.
  const __bf16* arow0 = he + (long)(off + m0 + (lane & 15)) * FE;
  const __bf16* arow1 = he + (long)(off + m0 + 16 + (lane & 15)) * FE;
  const __bf16* wd = ewd + (long)e * HDIM * FE;
  v8f acc[2][4] = {};
  for (int k0 = 0; k0 < FE; k0 += 32) {
    v16bf a0 = load_a_frag(arow0, k0, lane);
    v16bf a1 = load_a_frag(arow1, k0, lane);
#pragma unroll
    for (int j = 0; j < 4; ++j) {
      v16bf b = load_b_frag(wd, FE, n0 + j * 16, k0, lane);
      acc[0][j] = wmma_bf16(a0, b, acc[0][j]);
      acc[1][j] = wmma_bf16(a1, b, acc[1][j]);
    }
  }
  const int col = n0 + (lane & 15);
  const int radd = (lane >> 4) * 8;
#pragma unroll
  for (int mi = 0; mi < 2; ++mi)
#pragma unroll
    for (int j = 0; j < 4; ++j)
#pragma unroll
      for (int v = 0; v < 8; ++v) {
        const int m = m0 + mi * 16 + v + radd;
        if (m < cnt)
          eo[(long)(off + m) * HDIM + col + j * 16] = acc[mi][j][v];
      }
}

// out[t] = s_t*base[t] + w0*eo[slot0] + w1*eo[slot1]
__global__ __launch_bounds__(256) void combine_kernel(
    const float* __restrict__ baseo, const float* __restrict__ eo,
    const float* __restrict__ w0a, const float* __restrict__ w1a,
    const float* __restrict__ sarr, const int* __restrict__ slot0,
    const int* __restrict__ slot1, float* __restrict__ out) {
  const int t = blockIdx.x;
  const float s = sarr[t], w0 = w0a[t], w1 = w1a[t];
  const long b  = (long)t * HDIM;
  const long r0 = (long)slot0[t] * HDIM;
  const long r1 = (long)slot1[t] * HDIM;
#pragma unroll
  for (int i = 0; i < HDIM / 256; ++i) {
    const int h = threadIdx.x + i * 256;
    out[b + h] = s * baseo[b + h] + w0 * eo[r0 + h] + w1 * eo[r1 + h];
  }
}

// ---------------------------------------------------------------------------
extern "C" void kernel_launch(void* const* d_in, const int* in_sizes, int n_in,
                              void* d_out, int out_size, void* d_ws,
                              size_t ws_size, hipStream_t stream) {
  (void)in_sizes; (void)n_in; (void)out_size; (void)ws_size;
  const float* x    = (const float*)d_in[0];
  const float* bgw  = (const float*)d_in[1];
  const float* buw  = (const float*)d_in[2];
  const float* bdw  = (const float*)d_in[3];
  const float* rw   = (const float*)d_in[4];
  const float* egw  = (const float*)d_in[5];
  const float* euw  = (const float*)d_in[6];
  const float* edw  = (const float*)d_in[7];
  const float* alpha = (const float*)d_in[8];
  // d_in[9] = top_k (device scalar); fixed to 2 for this problem shape.
  float* out = (float*)d_out;

  char* ws = (char*)d_ws;
  size_t off = 0;
  auto walloc = [&](size_t bytes) -> void* {
    void* p = ws + off;
    off += (bytes + 255) & ~(size_t)255;
    return p;
  };

  __bf16* xb  = (__bf16*)walloc((size_t)T_TOK * HDIM * 2);
  __bf16* wgb = (__bf16*)walloc((size_t)FB * HDIM * 2);
  __bf16* wub = (__bf16*)walloc((size_t)FB * HDIM * 2);
  __bf16* wdb = (__bf16*)walloc((size_t)HDIM * FB * 2);
  __bf16* egb = (__bf16*)walloc((size_t)NEXP * FE * HDIM * 2);
  __bf16* eub = (__bf16*)walloc((size_t)NEXP * FE * HDIM * 2);
  __bf16* edb = (__bf16*)walloc((size_t)NEXP * HDIM * FE * 2);
  __bf16* hid = (__bf16*)walloc((size_t)T_TOK * FB * 2);
  float*  baseo = (float*)walloc((size_t)T_TOK * HDIM * 4);
  __bf16* he  = (__bf16*)walloc((size_t)(ENT + 32) * FE * 2);  // +32 row pad
  float*  eo  = (float*)walloc((size_t)ENT * HDIM * 4);
  float*  w0a = (float*)walloc((size_t)T_TOK * 4);
  float*  w1a = (float*)walloc((size_t)T_TOK * 4);
  float*  sarr = (float*)walloc((size_t)T_TOK * 4);
  int* idx0  = (int*)walloc((size_t)T_TOK * 4);
  int* idx1  = (int*)walloc((size_t)T_TOK * 4);
  int* slot0 = (int*)walloc((size_t)T_TOK * 4);
  int* slot1 = (int*)walloc((size_t)T_TOK * 4);
  int* list  = (int*)walloc((size_t)ENT * 4);
  int* counts  = (int*)walloc(NEXP * 4);
  int* offs    = (int*)walloc(NEXP * 4);
  int* cursors = (int*)walloc(NEXP * 4);

  zero_counts_kernel<<<1, 32, 0, stream>>>(counts);

  auto cvt = [&](const float* s, __bf16* d, long n) {
    long n4 = n / 4;
    int blocks = (int)((n4 + 255) / 256);
    if (blocks > 4096) blocks = 4096;
    cvt_f32_bf16<<<blocks, 256, 0, stream>>>(s, d, n4);
  };
  cvt(x,   xb,  (long)T_TOK * HDIM);
  cvt(bgw, wgb, (long)FB * HDIM);
  cvt(buw, wub, (long)FB * HDIM);
  cvt(bdw, wdb, (long)HDIM * FB);
  cvt(egw, egb, (long)NEXP * FE * HDIM);
  cvt(euw, eub, (long)NEXP * FE * HDIM);
  cvt(edw, edb, (long)NEXP * HDIM * FE);

  router_kernel<<<T_TOK / 8, 256, 0, stream>>>(x, rw, alpha, idx0, idx1, w0a,
                                               w1a, sarr, counts);
  scan_offsets_kernel<<<1, 32, 0, stream>>>(counts, offs, cursors);
  assign_slots_kernel<<<T_TOK / 256, 256, 0, stream>>>(idx0, idx1, offs,
                                                       cursors, list, slot0,
                                                       slot1);

  base_gateup_kernel<<<(T_TOK / 32) * (FB / 64) / 8, 256, 0, stream>>>(
      xb, wgb, wub, hid);
  base_down_kernel<<<(T_TOK / 32) * (HDIM / 64) / 8, 256, 0, stream>>>(
      hid, wdb, baseo);
  expert_gateup_kernel<<<NEXP * (T_TOK / 32) * (FE / 64) / 8, 256, 0, stream>>>(
      xb, egb, eub, list, counts, offs, he);
  expert_down_kernel<<<NEXP * (T_TOK / 32) * (HDIM / 64) / 8, 256, 0, stream>>>(
      he, edb, counts, offs, eo);
  combine_kernel<<<T_TOK, 256, 0, stream>>>(baseo, eo, w0a, w1a, sarr, slot0,
                                            slot1, out);
}
